// EmbeddingNetwork_34514357191319
// MI455X (gfx1250) — matrix operand order repaired and checked by reading, not compiled, not run on hardware
//
#include <hip/hip_runtime.h>

#define N_NODES 50000
#define N_EDGES 600000
#define F 128

typedef __attribute__((ext_vector_type(16))) _Float16 v16h;
typedef __attribute__((ext_vector_type(8)))  _Float16 v8h;
typedef __attribute__((ext_vector_type(8)))  float    v8f;

union U16 { v16h h; unsigned u[8]; };

static __device__ __forceinline__ unsigned pkh(float a, float b) {
    // v_cvt_pk_rtz_f16_f32: two f32 -> packed f16x2 in one VALU op
    return __builtin_bit_cast(unsigned, __builtin_amdgcn_cvt_pkrtz(a, b));
}

// ---------------------------------------------------------------------------
// Zero an int buffer
// ---------------------------------------------------------------------------
__global__ void zero_i32(int* __restrict__ p, int n) {
    int i = blockIdx.x * blockDim.x + threadIdx.x;
    if (i < n) p[i] = 0;
}

// ---------------------------------------------------------------------------
// CSR build step 1: per-node in-degree counts (int)
// ---------------------------------------------------------------------------
__global__ void count_kernel(const int* __restrict__ dst, int* __restrict__ counts) {
    int e = blockIdx.x * blockDim.x + threadIdx.x;
    if (e < N_EDGES) atomicAdd(&counts[dst[e]], 1);
}

// ---------------------------------------------------------------------------
// CSR build step 2: exclusive prefix sum over counts -> start[N+1].
// Single block of 256 threads; each thread scans a contiguous chunk.
// ---------------------------------------------------------------------------
__global__ void scan_kernel(const int* __restrict__ counts, int* __restrict__ start) {
    __shared__ int part[256];
    const int t = threadIdx.x;
    const int chunk = (N_NODES + 255) / 256;   // 196
    int s = 0;
    for (int j = 0; j < chunk; ++j) {
        int idx = t * chunk + j;
        if (idx < N_NODES) s += counts[idx];
    }
    part[t] = s;
    __syncthreads();
    // Hillis-Steele inclusive scan over the 256 partials
    for (int off = 1; off < 256; off <<= 1) {
        int v = (t >= off) ? part[t - off] : 0;
        __syncthreads();
        part[t] += v;
        __syncthreads();
    }
    int base = (t == 0) ? 0 : part[t - 1];
    for (int j = 0; j < chunk; ++j) {
        int idx = t * chunk + j;
        if (idx < N_NODES) { start[idx] = base; base += counts[idx]; }
    }
    if (t == 255) start[N_NODES] = base;
}

// ---------------------------------------------------------------------------
// CSR build step 3: bucket-fill edge source ids by destination node.
// ---------------------------------------------------------------------------
__global__ void fill_kernel(const int* __restrict__ src, const int* __restrict__ dst,
                            const int* __restrict__ start, int* __restrict__ cursor,
                            int* __restrict__ csr_src) {
    int e = blockIdx.x * blockDim.x + threadIdx.x;
    if (e >= N_EDGES) return;
    int d = dst[e];
    int pos = start[d] + atomicAdd(&cursor[d], 1);
    csr_src[pos] = src[e];
}

// ---------------------------------------------------------------------------
// Neighbor mean as a pure gather (no atomics): one wave per destination node.
// Each edge's 512B source row is read fully coalesced across 32 lanes
// (float4 per lane); sums stay in registers; one store per node.
// ---------------------------------------------------------------------------
__global__ void __launch_bounds__(256)
aggregate_kernel(const float* __restrict__ x, const int* __restrict__ csr_src,
                 const int* __restrict__ start, float* __restrict__ mean) {
    const int node = blockIdx.x * (blockDim.x >> 5) + (threadIdx.x >> 5);
    const int ln   = threadIdx.x & 31;
    if (node >= N_NODES) return;
    const int beg = start[node];
    const int end = start[node + 1];
    float ax = 0.f, ay = 0.f, az = 0.f, aw = 0.f;
    int i = beg;
    for (; i + 1 < end; i += 2) {          // 2-wide to keep more loads in flight
        int s0 = csr_src[i];
        int s1 = csr_src[i + 1];
        float4 v0 = *(const float4*)(x + (long)s0 * F + ln * 4);
        float4 v1 = *(const float4*)(x + (long)s1 * F + ln * 4);
        ax += v0.x; ay += v0.y; az += v0.z; aw += v0.w;
        ax += v1.x; ay += v1.y; az += v1.z; aw += v1.w;
    }
    if (i < end) {
        int s0 = csr_src[i];
        float4 v0 = *(const float4*)(x + (long)s0 * F + ln * 4);
        ax += v0.x; ay += v0.y; az += v0.z; aw += v0.w;
    }
    const float inv = 1.0f / fmaxf((float)(end - beg), 1.0f);
    float4 o = make_float4(ax * inv, ay * inv, az * inv, aw * inv);
    *(float4*)(mean + (long)node * F + ln * 4) = o;
}

// ---------------------------------------------------------------------------
// Pack all 6 weight matrices into f16, TRANSPOSED (column-major: wt[n*F+k]).
// Then every WMMA B operand is 16 contiguous halves (32B) -> global_load_b128.
// ---------------------------------------------------------------------------
__global__ void pack_weights(const float* __restrict__ w0, const float* __restrict__ w1,
                             const float* __restrict__ w2, const float* __restrict__ w3,
                             const float* __restrict__ w4, const float* __restrict__ w5,
                             _Float16* __restrict__ wt) {
    const int per_mat = F * F / 8;                 // 2048 stores per matrix
    int t = blockIdx.x * blockDim.x + threadIdx.x;
    if (t >= 6 * per_mat) return;
    int mat = t / per_mat;
    int r   = t - mat * per_mat;
    int n   = r >> 4;            // F/8 = 16 chunks per column
    int k0  = (r & 15) * 8;
    const float* w = (mat == 0) ? w0 : (mat == 1) ? w1 : (mat == 2) ? w2
                   : (mat == 3) ? w3 : (mat == 4) ? w4 : w5;
    v8h o;
#pragma unroll
    for (int j = 0; j < 8; ++j) o[j] = (_Float16)w[(long)(k0 + j) * F + n];
    *(v8h*)(wt + (long)mat * F * F + (long)n * F + k0) = o;
}

// ---------------------------------------------------------------------------
// Fused SAGE dense part: out = [relu]( x@Wself + mean@Wneigh + b )
// One wave computes a 16-row tile of out (16 x 128); 64 v_wmma per wave.
// wtS/wtN are f16, column-major (packed by pack_weights). No LDS, no barriers.
// ---------------------------------------------------------------------------
__global__ void __launch_bounds__(256)
sage_wmma(const float* __restrict__ x, const float* __restrict__ mean,
          const _Float16* __restrict__ wtS, const _Float16* __restrict__ wtN,
          const float* __restrict__ bias, float* __restrict__ out, int relu)
{
    const int wave  = threadIdx.x >> 5;
    const int lane  = threadIdx.x & 31;
    const int lrow  = lane & 15;   // row within A / col within B,C,D
    const int lhalf = lane >> 4;   // lane-half select
    const int row0  = (blockIdx.x * (blockDim.x >> 5) + wave) * 16;
    if (row0 >= N_NODES) return;   // wave-uniform: EXEC stays all-ones for WMMA

    const int row = row0 + lrow;
    const float* xr = x + (long)row * F;
    const float* mr = mean + (long)row * F;

    // Build A operands (16x32 f16 chunks) for self (x) and neighbor (mean).
    // A layout: element e (0..15) of lane (lhalf,lrow) holds
    //   k = kc*32 + lhalf*16 + e  -> [base..base+15]
    // -> four aligned float4 loads per chunk per matrix, packed via cvt_pk.
    U16 axs[4], axn[4];
#pragma unroll
    for (int kc = 0; kc < 4; ++kc) {
        const int base = kc * 32 + lhalf * 16;
        float4 x0 = *(const float4*)(xr + base);
        float4 x1 = *(const float4*)(xr + base + 4);
        float4 x2 = *(const float4*)(xr + base + 8);
        float4 x3 = *(const float4*)(xr + base + 12);
        axs[kc].u[0] = pkh(x0.x, x0.y); axs[kc].u[1] = pkh(x0.z, x0.w);
        axs[kc].u[2] = pkh(x1.x, x1.y); axs[kc].u[3] = pkh(x1.z, x1.w);
        axs[kc].u[4] = pkh(x2.x, x2.y); axs[kc].u[5] = pkh(x2.z, x2.w);
        axs[kc].u[6] = pkh(x3.x, x3.y); axs[kc].u[7] = pkh(x3.z, x3.w);
        float4 m0 = *(const float4*)(mr + base);
        float4 m1 = *(const float4*)(mr + base + 4);
        float4 m2 = *(const float4*)(mr + base + 8);
        float4 m3 = *(const float4*)(mr + base + 12);
        axn[kc].u[0] = pkh(m0.x, m0.y); axn[kc].u[1] = pkh(m0.z, m0.w);
        axn[kc].u[2] = pkh(m1.x, m1.y); axn[kc].u[3] = pkh(m1.z, m1.w);
        axn[kc].u[4] = pkh(m2.x, m2.y); axn[kc].u[5] = pkh(m2.z, m2.w);
        axn[kc].u[6] = pkh(m3.x, m3.y); axn[kc].u[7] = pkh(m3.z, m3.w);
    }

    // 8 output column tiles of 16; K = 128 in 4 chunks of 32; 2 matmuls each.
    // Accumulators start at zero (WMMA C can fold to inline 0); bias in epilogue.
#pragma unroll
    for (int nt = 0; nt < 8; ++nt) {
        const int n = nt * 16 + lrow;
        v8f acc = { 0.f, 0.f, 0.f, 0.f, 0.f, 0.f, 0.f, 0.f };
#pragma unroll
        for (int kc = 0; kc < 4; ++kc) {
            // B (32x16) layout: lane (lhalf,lrow) holds column n, K = kc*32 +
            // lhalf*16 + (0..15) -> 16 contiguous halves in column-major wt.
            const long boff = (long)n * F + kc * 32 + lhalf * 16;
            v16h bs = *(const v16h*)(wtS + boff);
            v16h bn = *(const v16h*)(wtN + boff);
            acc = __builtin_amdgcn_wmma_f32_16x16x32_f16(false, axs[kc].h, false, bs,
                                                         (short)0, acc, false, false);
            acc = __builtin_amdgcn_wmma_f32_16x16x32_f16(false, axn[kc].h, false, bn,
                                                         (short)0, acc, false, false);
        }
        // C/D layout: VGPR r holds (M = r + 8*lhalf, N = lrow) of the 16x16 tile
        const float bv = bias[n];
#pragma unroll
        for (int r = 0; r < 8; ++r) {
            float v = acc[r] + bv;
            if (relu) v = fmaxf(v, 0.0f);
            out[(long)(row0 + lhalf * 8 + r) * F + n] = v;
        }
    }
}

// ---------------------------------------------------------------------------
extern "C" void kernel_launch(void* const* d_in, const int* in_sizes, int n_in,
                              void* d_out, int out_size, void* d_ws, size_t ws_size,
                              hipStream_t stream) {
    const float* features = (const float*)d_in[0];
    const int*   src      = (const int*)d_in[1];
    const int*   dst      = (const int*)d_in[2];
    const float* W1s = (const float*)d_in[3];
    const float* W1n = (const float*)d_in[4];
    const float* b1  = (const float*)d_in[5];
    const float* W2s = (const float*)d_in[6];
    const float* W2n = (const float*)d_in[7];
    const float* b2  = (const float*)d_in[8];
    const float* W3s = (const float*)d_in[9];
    const float* W3n = (const float*)d_in[10];
    const float* b3  = (const float*)d_in[11];
    float* out = (float*)d_out;

    // Workspace layout
    float*    mean    = (float*)d_ws;                         // N*F f32
    float*    hbuf    = mean + (size_t)N_NODES * F;           // N*F f32 (h2)
    _Float16* wt      = (_Float16*)(hbuf + (size_t)N_NODES * F); // 6*F*F f16
    int*      counts  = (int*)(wt + 6L * F * F);              // N
    int*      start   = counts + N_NODES;                     // N+1
    int*      cursor  = start + N_NODES + 1;                  // N
    int*      csr_src = cursor + N_NODES;                     // E

    const dim3 blk(256);
    const int eb = (N_EDGES + 255) / 256;       // edge-parallel blocks
    const int nb = (N_NODES + 255) / 256;       // node-parallel blocks
    const int wb = (N_NODES / 16 + 7) / 8;      // 1 wave per 16-row tile
    const int ab = (N_NODES + 7) / 8;           // 1 wave per node (8 waves/block)

    // Pack weights (f16, transposed) once per call
    pack_weights<<<(6 * F * F / 8 + 255) / 256, blk, 0, stream>>>(
        W1s, W1n, W2s, W2n, W3s, W3n, wt);
    const _Float16* wt1s = wt + 0L * F * F;
    const _Float16* wt1n = wt + 1L * F * F;
    const _Float16* wt2s = wt + 2L * F * F;
    const _Float16* wt2n = wt + 3L * F * F;
    const _Float16* wt3s = wt + 4L * F * F;
    const _Float16* wt3n = wt + 5L * F * F;

    // Build CSR (dst -> list of src) once; shared by all three layers.
    zero_i32<<<nb, blk, 0, stream>>>(counts, N_NODES);
    count_kernel<<<eb, blk, 0, stream>>>(dst, counts);
    scan_kernel<<<1, blk, 0, stream>>>(counts, start);
    zero_i32<<<nb, blk, 0, stream>>>(cursor, N_NODES);
    fill_kernel<<<eb, blk, 0, stream>>>(src, dst, start, cursor, csr_src);

    // Layer 1: x = features -> h1 in d_out (relu)
    aggregate_kernel<<<ab, blk, 0, stream>>>(features, csr_src, start, mean);
    sage_wmma<<<wb, blk, 0, stream>>>(features, mean, wt1s, wt1n, b1, out, 1);

    // Layer 2: x = h1 (d_out) -> h2 in hbuf (relu)
    aggregate_kernel<<<ab, blk, 0, stream>>>(out, csr_src, start, mean);
    sage_wmma<<<wb, blk, 0, stream>>>(out, mean, wt2s, wt2n, b2, hbuf, 1);

    // Layer 3: x = h2 (hbuf) -> d_out (no relu)
    aggregate_kernel<<<ab, blk, 0, stream>>>(hbuf, csr_src, start, mean);
    sage_wmma<<<wb, blk, 0, stream>>>(hbuf, mean, wt3s, wt3n, b3, out, 0);
}
